// NNConvModel_54494545052434
// MI455X (gfx1250) — compile-verified, matrix operand order checked by reading, not run on hardware
//
#include <hip/hip_runtime.h>
#include <hip/hip_bf16.h>

typedef __attribute__((ext_vector_type(16))) _Float16 v16h;
typedef __attribute__((ext_vector_type(8)))  _Float16 v8h;
typedef __attribute__((ext_vector_type(8)))  float    v8f;

#define NND 50000
#define NED 200000
#define HD  32
#define BGR 128
#define HSTRIDE 132   // 128 MLP hidden + 1 bias-one + pad

static inline int cdiv_h(int a, int b) { return (a + b - 1) / b; }

// ---------------- weight packing: B-fragment layout (ISA 7.12.2) -------------
// frag[((tc*32)+lane)*16 + j] = W[k, tc*16 + (lane&15)]
// k = ((j<8)?0:16) + ((lane<16)?0:8) + (j&7)
__global__ void k_pack_bfrag(const float* __restrict__ W, _Float16* __restrict__ frag, int ncols, int total) {
    int idx = blockIdx.x * blockDim.x + threadIdx.x;
    if (idx >= total) return;
    int j = idx & 15, l = (idx >> 4) & 31, tc = idx >> 9;
    int k = ((j < 8) ? 0 : 16) + ((l < 16) ? 0 : 8) + (j & 7);
    int c = tc * 16 + (l & 15);
    frag[idx] = (_Float16)W[k * ncols + c];
}

// nn2 rearranged fragments: for K-step t (0..128), tile tc (0,1):
// value = (t<128) ? nn2_w[t*1024 + h*32 + o] : nn2_b[h*32 + o]
__global__ void k_pack_nn2(const float* __restrict__ nn2w, const float* __restrict__ nn2b,
                           _Float16* __restrict__ frag, int total) {
    int idx = blockIdx.x * blockDim.x + threadIdx.x;
    if (idx >= total) return;
    int j = idx & 15, l = (idx >> 4) & 31, tc = (idx >> 9) & 1, t = idx >> 10;
    int h = ((j < 8) ? 0 : 16) + ((l < 16) ? 0 : 8) + (j & 7);
    int o = tc * 16 + (l & 15);
    float v = (t < 128) ? nn2w[t * 1024 + h * 32 + o] : nn2b[h * 32 + o];
    frag[idx] = (_Float16)v;
}

// ---------------- edge MLP layer 1 (K=3, pure VALU) --------------------------
__global__ void k_edge_hidden(const float* __restrict__ ea, const float* __restrict__ w1,
                              const float* __restrict__ b1, _Float16* __restrict__ hid, int total) {
    int idx = blockIdx.x * blockDim.x + threadIdx.x;
    if (idx >= total) return;
    int e = idx / HSTRIDE, t = idx % HSTRIDE;
    float v = 0.f;
    if (t < 128) {
        v = b1[t] + ea[e*3+0]*w1[t] + ea[e*3+1]*w1[128+t] + ea[e*3+2]*w1[256+t];
        v = fmaxf(v, 0.f);
    } else if (t == 128) v = 1.f;
    hid[idx] = (_Float16)v;
}

__global__ void k_f32_to_h16(const float* __restrict__ src, _Float16* __restrict__ dst, int n) {
    int idx = blockIdx.x * blockDim.x + threadIdx.x;
    if (idx < n) dst[idx] = (_Float16)src[idx];
}

__global__ void k_deg(const int* __restrict__ ei, float* __restrict__ deg, int ne) {
    int e = blockIdx.x * blockDim.x + threadIdx.x;
    if (e < ne) atomicAdd(&deg[ei[ne + e]], 1.0f);
}

__global__ void k_agg_norm(float* __restrict__ agg, const float* __restrict__ deg, int total) {
    int idx = blockIdx.x * blockDim.x + threadIdx.x;
    if (idx < total) agg[idx] /= fmaxf(deg[idx >> 5], 1.0f);
}

// ---------------- NNConv message: fused bilinear GEMM via WMMA ---------------
// msg[E,32] = Z[E,4128] x nn2hat[4128,32]; A row = hidden[e,:] (x) out[src[e],:]
// K-tile t == MLP hidden index m, so A(step t) = (preloaded out-frag) * hidden[e,t]
__global__ void k_msg(const _Float16* __restrict__ stateh, const _Float16* __restrict__ hid,
                      const _Float16* __restrict__ nn2frag, const int* __restrict__ ei,
                      float* __restrict__ agg, int nwave) {
    int wv = (blockIdx.x * blockDim.x + threadIdx.x) >> 5;
    if (wv >= nwave) return;                 // wave-uniform: EXEC stays all-ones
    int lane = threadIdx.x & 31;
    int row  = lane & 15;
    int kgrp = (lane < 16) ? 0 : 8;
    int e = wv * 16 + row;
    int s = ei[e];
    const _Float16* srow = stateh + s * 32;
    v8h lo = *(const v8h*)(srow + kgrp);
    v8h hi = *(const v8h*)(srow + 16 + kgrp);
    v16h av;
#pragma unroll
    for (int i = 0; i < 8; ++i) { av[i] = lo[i]; av[8 + i] = hi[i]; }
    v8f c0 = {}; v8f c1 = {};
    const _Float16* hrow = hid + e * HSTRIDE;
#pragma unroll 3
    for (int t = 0; t < 129; ++t) {
        _Float16 hv = hrow[t];
        v16h a;
#pragma unroll
        for (int i = 0; i < 16; ++i) a[i] = av[i] * hv;
        const _Float16* bp = nn2frag + ((t * 2) * 32 + lane) * 16;
        v16h b0 = *(const v16h*)bp;
        v16h b1 = *(const v16h*)(bp + 32 * 16);
        c0 = __builtin_amdgcn_wmma_f32_16x16x32_f16(false, a, false, b0, (short)0, c0, false, false);
        c1 = __builtin_amdgcn_wmma_f32_16x16x32_f16(false, a, false, b1, (short)0, c1, false, false);
    }
    int o = lane & 15;
    int mbase = (lane < 16) ? 0 : 8;
#pragma unroll
    for (int i = 0; i < 8; ++i) {
        int em = wv * 16 + mbase + i;
        int d  = ei[NED + em];
        atomicAdd(&agg[d * 32 + o],      c0[i]);
        atomicAdd(&agg[d * 32 + 16 + o], c1[i]);
    }
}

// ---------------- generic [rows,32] x [32,ncols] WMMA GEMM -------------------
__global__ void k_gemm32(const _Float16* __restrict__ A, const _Float16* __restrict__ Bfrag,
                         const float* __restrict__ bias, const float* __restrict__ add,
                         float* __restrict__ C, int nwave, int ncols, int dorelu) {
    int wv = (blockIdx.x * blockDim.x + threadIdx.x) >> 5;
    if (wv >= nwave) return;
    int lane = threadIdx.x & 31;
    int kgrp = (lane < 16) ? 0 : 8;
    int rowbase = wv * 16;
    const _Float16* arow = A + (rowbase + (lane & 15)) * 32;
    v8h lo = *(const v8h*)(arow + kgrp);
    v8h hi = *(const v8h*)(arow + 16 + kgrp);
    v16h a;
#pragma unroll
    for (int i = 0; i < 8; ++i) { a[i] = lo[i]; a[8 + i] = hi[i]; }
    int ntiles = ncols >> 4;
    int o = lane & 15, mbase = (lane < 16) ? 0 : 8;
    for (int tc = 0; tc < ntiles; ++tc) {
        v16h b = *(const v16h*)(Bfrag + (tc * 32 + lane) * 16);
        v8f c = {};
        c = __builtin_amdgcn_wmma_f32_16x16x32_f16(false, a, false, b, (short)0, c, false, false);
        int col = tc * 16 + o;
        float bb = bias[col];
#pragma unroll
        for (int i = 0; i < 8; ++i) {
            int r = rowbase + mbase + i;
            float v = c[i] + bb;
            if (add) v += add[r * ncols + col];
            if (dorelu) v = fmaxf(v, 0.f);
            C[r * ncols + col] = v;
        }
    }
}

// ---------------- GRU gate elementwise ---------------------------------------
__device__ __forceinline__ float sigf(float x) { return 1.f / (1.f + __expf(-x)); }

__global__ void k_gru(const float* __restrict__ gx, const float* __restrict__ gh,
                      float* __restrict__ state, int total) {
    int idx = blockIdx.x * blockDim.x + threadIdx.x;
    if (idx >= total) return;
    int n = idx >> 5, d = idx & 31;
    float xr = gx[n*96 + d], xz = gx[n*96 + 32 + d], xn = gx[n*96 + 64 + d];
    float hr = gh[n*96 + d], hz = gh[n*96 + 32 + d], hn = gh[n*96 + 64 + d];
    float r = sigf(xr + hr);
    float z = sigf(xz + hz);
    float nc = tanhf(xn + r * hn);
    state[idx] = (1.f - z) * nc + z * state[idx];
}

// ---------------- Set2Set ----------------------------------------------------
__device__ __forceinline__ unsigned enc_ord(float f) {
    unsigned u = __float_as_uint(f);
    return (u & 0x80000000u) ? ~u : (u | 0x80000000u);
}
__device__ __forceinline__ float dec_ord(unsigned u) {
    return (u & 0x80000000u) ? __uint_as_float(u & 0x7fffffffu) : __uint_as_float(~u);
}

__global__ void k_lstm_g(const float* __restrict__ qs, const float* __restrict__ hs,
                         const float* __restrict__ wih, const float* __restrict__ whh,
                         const float* __restrict__ bih, const float* __restrict__ bhh,
                         float* __restrict__ g) {
    int idx = blockIdx.x * blockDim.x + threadIdx.x;
    if (idx >= BGR * 128) return;
    int b = idx >> 7, j = idx & 127;
    float acc = bih[j] + bhh[j];
#pragma unroll 4
    for (int k = 0; k < 64; ++k) acc += qs[b*64 + k] * wih[k*128 + j];
#pragma unroll 4
    for (int k = 0; k < 32; ++k) acc += hs[b*32 + k] * whh[k*128 + j];
    g[idx] = acc;
}

__global__ void k_lstm_gate(const float* __restrict__ g, float* __restrict__ cs, float* __restrict__ hs) {
    int idx = blockIdx.x * blockDim.x + threadIdx.x;
    if (idx >= BGR * 32) return;
    int b = idx >> 5, d = idx & 31;
    float gi = g[b*128 + d], gf = g[b*128 + 32 + d], gg = g[b*128 + 64 + d], go = g[b*128 + 96 + d];
    float c = sigf(gf) * cs[idx] + sigf(gi) * tanhf(gg);
    cs[idx] = c;
    hs[idx] = sigf(go) * tanhf(c);
}

__global__ void k_init_step(unsigned* __restrict__ emax, float* __restrict__ asum, float* __restrict__ rr) {
    int idx = blockIdx.x * blockDim.x + threadIdx.x;
    if (idx >= BGR * 32) return;
    rr[idx] = 0.f;
    if ((idx & 31) == 0) { emax[idx >> 5] = 0u; asum[idx >> 5] = 0.f; }
}

__global__ void k_escore(const float* __restrict__ ne, const float* __restrict__ hs,
                         const int* __restrict__ batch, float* __restrict__ es, unsigned* __restrict__ emax) {
    int n = blockIdx.x * blockDim.x + threadIdx.x;
    if (n >= NND) return;
    int b = batch[n];
    float dot = 0.f;
#pragma unroll 8
    for (int d = 0; d < 32; ++d) dot += ne[n*32 + d] * hs[b*32 + d];
    es[n] = dot;
    atomicMax(&emax[b], enc_ord(dot));
}

__global__ void k_attn(const float* __restrict__ es, const unsigned* __restrict__ emax,
                       const int* __restrict__ batch, float* __restrict__ aw, float* __restrict__ asum) {
    int n = blockIdx.x * blockDim.x + threadIdx.x;
    if (n >= NND) return;
    int b = batch[n];
    float v = __expf(es[n] - dec_ord(emax[b]));
    aw[n] = v;
    atomicAdd(&asum[b], v);
}

__global__ void k_read(const float* __restrict__ ne, const float* __restrict__ aw,
                       const float* __restrict__ asum, const int* __restrict__ batch,
                       float* __restrict__ rr, int total) {
    int idx = blockIdx.x * blockDim.x + threadIdx.x;
    if (idx >= total) return;
    int n = idx >> 5, d = idx & 31;
    int b = batch[n];
    float s = asum[b];
    float w = aw[n] / ((s == 0.f) ? 1.f : s);
    atomicAdd(&rr[b*32 + d], w * ne[idx]);
}

__global__ void k_qstar(const float* __restrict__ hs, const float* __restrict__ rr, float* __restrict__ qs) {
    int idx = blockIdx.x * blockDim.x + threadIdx.x;
    if (idx >= BGR * 64) return;
    int b = idx >> 6, k = idx & 63;
    qs[idx] = (k < 32) ? hs[b*32 + k] : rr[b*32 + (k - 32)];
}

__global__ void k_graph(const float* __restrict__ qs, const float* __restrict__ w2,
                        const float* __restrict__ b2, float* __restrict__ ge) {
    int idx = blockIdx.x * blockDim.x + threadIdx.x;
    if (idx >= BGR * 32) return;
    int b = idx >> 5, d = idx & 31;
    float acc = b2[d];
#pragma unroll 4
    for (int k = 0; k < 64; ++k) acc += qs[b*64 + k] * w2[k*32 + d];
    ge[idx] = acc;
}

__global__ void k_pred(const float* __restrict__ ge, const float* __restrict__ w3,
                       const float* __restrict__ b3, float* __restrict__ pred) {
    int b = blockIdx.x * blockDim.x + threadIdx.x;
    if (b >= BGR) return;
    float acc = b3[0];
#pragma unroll 8
    for (int d = 0; d < 32; ++d) acc += ge[b*32 + d] * w3[d];
    pred[b] = acc;
}

// ---------------- host orchestration -----------------------------------------
extern "C" void kernel_launch(void* const* d_in, const int* in_sizes, int n_in,
                              void* d_out, int out_size, void* d_ws, size_t ws_size,
                              hipStream_t stream) {
    const float* x       = (const float*)d_in[0];
    const float* ea      = (const float*)d_in[1];
    const float* lin0_w  = (const float*)d_in[2];
    const float* lin0_b  = (const float*)d_in[3];
    const float* nn1_w   = (const float*)d_in[4];
    const float* nn1_b   = (const float*)d_in[5];
    const float* nn2_w   = (const float*)d_in[6];
    const float* nn2_b   = (const float*)d_in[7];
    const float* root_w  = (const float*)d_in[8];
    const float* conv_b  = (const float*)d_in[9];
    const float* gwih    = (const float*)d_in[10];
    const float* gwhh    = (const float*)d_in[11];
    const float* gbih    = (const float*)d_in[12];
    const float* gbhh    = (const float*)d_in[13];
    const float* lin1_w  = (const float*)d_in[14];
    const float* lin1_b  = (const float*)d_in[15];
    const float* lwih    = (const float*)d_in[16];
    const float* lwhh    = (const float*)d_in[17];
    const float* lbih    = (const float*)d_in[18];
    const float* lbhh    = (const float*)d_in[19];
    const float* lin2_w  = (const float*)d_in[20];
    const float* lin2_b  = (const float*)d_in[21];
    const float* lin3_w  = (const float*)d_in[22];
    const float* lin3_b  = (const float*)d_in[23];
    const int*   ei      = (const int*)d_in[24];
    const int*   batch   = (const int*)d_in[25];

    float* out = (float*)d_out;
    float* pred = out;                       // [128]
    float* ge   = out + BGR;                 // [128,32]
    float* ne   = out + BGR + BGR*32;        // [N,32] node_embed

    // workspace carve (256B aligned)
    char* p = (char*)d_ws;
    auto carve = [&](size_t bytes) { char* r = p; p += (bytes + 255) & ~(size_t)255; return (void*)r; };
    _Float16* hid      = (_Float16*)carve((size_t)NED * HSTRIDE * 2);
    _Float16* nn2frag  = (_Float16*)carve((size_t)129 * 2 * 32 * 16 * 2);
    _Float16* f_lin0   = (_Float16*)carve(2 * 32 * 16 * 2);
    _Float16* f_root   = (_Float16*)carve(2 * 32 * 16 * 2);
    _Float16* f_wih    = (_Float16*)carve(6 * 32 * 16 * 2);
    _Float16* f_whh    = (_Float16*)carve(6 * 32 * 16 * 2);
    _Float16* f_lin1   = (_Float16*)carve(2 * 32 * 16 * 2);
    float*    state    = (float*)carve((size_t)NND * 32 * 4);
    _Float16* stateh   = (_Float16*)carve((size_t)NND * 32 * 2);
    _Float16* bufAh    = (_Float16*)carve((size_t)NND * 32 * 2);
    float*    agg      = (float*)carve((size_t)NND * 32 * 4);
    float*    mbuf     = (float*)carve((size_t)NND * 32 * 4);
    float*    gx       = (float*)carve((size_t)NND * 96 * 4);
    float*    gh       = (float*)carve((size_t)NND * 96 * 4);
    float*    deg      = (float*)carve((size_t)NND * 4);
    float*    es       = (float*)carve((size_t)NND * 4);
    float*    aw       = (float*)carve((size_t)NND * 4);
    float*    glstm    = (float*)carve(BGR * 128 * 4);
    float*    qstar    = (float*)carve(BGR * 64 * 4);
    float*    hsb      = (float*)carve(BGR * 32 * 4);
    float*    csb      = (float*)carve(BGR * 32 * 4);
    float*    asum     = (float*)carve(BGR * 4);
    unsigned* emax     = (unsigned*)carve(BGR * 4);
    float*    rr       = (float*)carve(BGR * 32 * 4);

    const int TPB = 256;
    // --- pack weights into WMMA B-fragment order (fp16) ---
    k_pack_bfrag<<<cdiv_h(32*32, TPB),  TPB, 0, stream>>>(lin0_w, f_lin0, 32, 32*32);
    k_pack_bfrag<<<cdiv_h(32*32, TPB),  TPB, 0, stream>>>(root_w, f_root, 32, 32*32);
    k_pack_bfrag<<<cdiv_h(96*32, TPB),  TPB, 0, stream>>>(gwih,   f_wih,  96, 96*32);
    k_pack_bfrag<<<cdiv_h(96*32, TPB),  TPB, 0, stream>>>(gwhh,   f_whh,  96, 96*32);
    k_pack_bfrag<<<cdiv_h(32*32, TPB),  TPB, 0, stream>>>(lin1_w, f_lin1, 32, 32*32);
    k_pack_nn2<<<cdiv_h(129*1024, TPB), TPB, 0, stream>>>(nn2_w, nn2_b, nn2frag, 129*1024);

    // --- edge MLP layer 1 (fixed across iterations) + degree ---
    k_edge_hidden<<<cdiv_h(NED*HSTRIDE, TPB), TPB, 0, stream>>>(ea, nn1_w, nn1_b, hid, NED*HSTRIDE);
    hipMemsetAsync(deg, 0, (size_t)NND * 4, stream);
    k_deg<<<cdiv_h(NED, TPB), TPB, 0, stream>>>(ei, deg, NED);

    // --- lin0: state = relu(x @ lin0_w + b) via WMMA ---
    const int NW = NND / 16;                       // 3125 waves, exact
    k_f32_to_h16<<<cdiv_h(NND*32, TPB), TPB, 0, stream>>>(x, bufAh, NND*32);
    k_gemm32<<<cdiv_h(NW*32, TPB), TPB, 0, stream>>>(bufAh, f_lin0, lin0_b, nullptr, state, NW, 32, 1);
    k_f32_to_h16<<<cdiv_h(NND*32, TPB), TPB, 0, stream>>>(state, stateh, NND*32);

    // --- 3 message-passing + GRU iterations ---
    const int EW = NED / 16;                       // 12500 waves, exact
    for (int it = 0; it < 3; ++it) {
        hipMemsetAsync(agg, 0, (size_t)NND * 32 * 4, stream);
        k_msg<<<cdiv_h(EW*32, TPB), TPB, 0, stream>>>(stateh, hid, nn2frag, ei, agg, EW);
        k_agg_norm<<<cdiv_h(NND*32, TPB), TPB, 0, stream>>>(agg, deg, NND*32);
        // m = relu(state @ root_w + agg + conv_b)
        k_gemm32<<<cdiv_h(NW*32, TPB), TPB, 0, stream>>>(stateh, f_root, conv_b, agg, mbuf, NW, 32, 1);
        k_f32_to_h16<<<cdiv_h(NND*32, TPB), TPB, 0, stream>>>(mbuf, bufAh, NND*32);
        // GRU gate GEMMs
        k_gemm32<<<cdiv_h(NW*32, TPB), TPB, 0, stream>>>(bufAh,  f_wih, gbih, nullptr, gx, NW, 96, 0);
        k_gemm32<<<cdiv_h(NW*32, TPB), TPB, 0, stream>>>(stateh, f_whh, gbhh, nullptr, gh, NW, 96, 0);
        k_gru<<<cdiv_h(NND*32, TPB), TPB, 0, stream>>>(gx, gh, state, NND*32);
        k_f32_to_h16<<<cdiv_h(NND*32, TPB), TPB, 0, stream>>>(state, stateh, NND*32);
    }

    // --- node_embed = state @ lin1_w + b (into d_out) ---
    k_gemm32<<<cdiv_h(NW*32, TPB), TPB, 0, stream>>>(stateh, f_lin1, lin1_b, nullptr, ne, NW, 32, 0);

    // --- Set2Set, 3 steps ---
    hipMemsetAsync(qstar, 0, BGR * 64 * 4, stream);
    hipMemsetAsync(hsb,   0, BGR * 32 * 4, stream);
    hipMemsetAsync(csb,   0, BGR * 32 * 4, stream);
    for (int st = 0; st < 3; ++st) {
        k_lstm_g<<<cdiv_h(BGR*128, TPB), TPB, 0, stream>>>(qstar, hsb, lwih, lwhh, lbih, lbhh, glstm);
        k_lstm_gate<<<cdiv_h(BGR*32, TPB), TPB, 0, stream>>>(glstm, csb, hsb);
        k_init_step<<<cdiv_h(BGR*32, TPB), TPB, 0, stream>>>(emax, asum, rr);
        k_escore<<<cdiv_h(NND, TPB), TPB, 0, stream>>>(ne, hsb, batch, es, emax);
        k_attn<<<cdiv_h(NND, TPB), TPB, 0, stream>>>(es, emax, batch, aw, asum);
        k_read<<<cdiv_h(NND*32, TPB), TPB, 0, stream>>>(ne, aw, asum, batch, rr, NND*32);
        k_qstar<<<cdiv_h(BGR*64, TPB), TPB, 0, stream>>>(hsb, rr, qstar);
    }

    // --- readout ---
    k_graph<<<cdiv_h(BGR*32, TPB), TPB, 0, stream>>>(qstar, lin2_w, lin2_b, ge);
    k_pred<<<cdiv_h(BGR, TPB), TPB, 0, stream>>>(ge, lin3_w, lin3_b, pred);
}
